// MultiHeadAttention_25975962206789
// MI455X (gfx1250) — matrix operand order repaired and checked by reading, not compiled, not run on hardware
//
#include <hip/hip_runtime.h>

// MHA forward for MI455X (gfx1250, wave32, WMMA).
// B=2, S=2048, D=1024, H=16, hd=64.
// Pipeline: cvt/transpose -> QKV gemm (wmma f16) -> flash attention (wmma f16,
// online softmax in fp32, LDS transpose for P) -> out-proj gemm (wmma f16, fp32 out).
//
// Fragment layouts per CDNA5 ISA 7.12.2:
//  A (16x32 f16): lane L holds row M=L%16; lane half 0 -> K {0..7,16..23},
//                 half 1 -> K {8..15,24..31} (interleaved blocks of 8).
//  B (32x16 f16): lane L holds col N=L%16; lane half 0 -> K {0..7,16..23},
//                 half 1 -> K {8..15,24..31} (same interleave as A).
//  C/D (16x16 f32): element r of v8f -> M = r + 8*half, N = L%16.

#define BATCH    2
#define SEQ      2048
#define DIM      1024
#define NHEAD    16
#define HDIM     64
#define TOKENS   (BATCH * SEQ)        // 4096
#define QKV_N    (3 * DIM)            // 3072

typedef _Float16 v16h __attribute__((ext_vector_type(16)));
typedef _Float16 v8h  __attribute__((ext_vector_type(8)));
typedef float    v8f  __attribute__((ext_vector_type(8)));

union Frag { v16h v; v8h h[2]; };

__device__ __forceinline__ v8f vzero() {
    v8f z;
#pragma unroll
    for (int i = 0; i < 8; ++i) z[i] = 0.0f;
    return z;
}

// A-operand fragment: p points at (row, kchunk + half*8); second 8 f16 are
// 16 elements further on (interleaved K blocks of 8 per lane half).
__device__ __forceinline__ Frag load_frag_a(const _Float16* p) {
    Frag f;
    f.h[0] = *(const v8h*)(p);
    f.h[1] = *(const v8h*)(p + 16);
    return f;
}

// B-operand fragment: p points at (col, kchunk + half*8); second 8 f16 are
// 16 elements further on (interleaved K blocks of 8 per lane half).
__device__ __forceinline__ Frag load_frag_b(const _Float16* p) {
    Frag f;
    f.h[0] = *(const v8h*)(p);
    f.h[1] = *(const v8h*)(p + 16);
    return f;
}

__device__ __forceinline__ v8f wmma16(const Frag& a, const Frag& b, v8f c) {
    return __builtin_amdgcn_wmma_f32_16x16x32_f16(false, a.v, false, b.v,
                                                  (short)0, c, false, false);
}

__device__ __forceinline__ float rmax16(float v) {
#pragma unroll
    for (int o = 8; o >= 1; o >>= 1) v = fmaxf(v, __shfl_xor(v, o, 16));
    return v;
}
__device__ __forceinline__ float rsum16(float v) {
#pragma unroll
    for (int o = 8; o >= 1; o >>= 1) v += __shfl_xor(v, o, 16);
    return v;
}

// ---------------- preprocessing ----------------

__global__ void k_cvt_f16(const float* __restrict__ src, _Float16* __restrict__ dst, int n) {
    int i = blockIdx.x * blockDim.x + threadIdx.x;
    if (i < n) dst[i] = (_Float16)src[i];
}

// src[K][N] fp32 (row-major)  ->  dst[N][K] f16 (weights to column-major)
__global__ void k_transpose_cvt(const float* __restrict__ src, _Float16* __restrict__ dst,
                                int K, int N) {
    int i = blockIdx.x * blockDim.x + threadIdx.x;
    if (i < K * N) {
        int n = i / K;
        int k = i - n * K;
        dst[i] = (_Float16)src[(size_t)k * N + n];
    }
}

// ---------------- QKV GEMM ----------------
// C[4096][3072] = x16[4096][1024] @ WqkvT^T + b ; scatter into Q,K (row-major
// per head) and V (transposed per head). One wave per 64x64 tile.
__global__ void __launch_bounds__(32)
k_gemm_qkv(const _Float16* __restrict__ A, const _Float16* __restrict__ Bt,
           const float* __restrict__ bias,
           _Float16* __restrict__ q16, _Float16* __restrict__ k16,
           _Float16* __restrict__ vT) {
    const int lane = threadIdx.x & 31;
    const int half = lane >> 4;
    const int lr   = lane & 15;
    const int nbase = blockIdx.x * 64;
    const int mbase = blockIdx.y * 64;

    v8f acc[4][4];
#pragma unroll
    for (int i = 0; i < 4; ++i)
#pragma unroll
        for (int j = 0; j < 4; ++j) acc[i][j] = vzero();

    for (int k0 = 0; k0 < DIM; k0 += 32) {
        Frag af[4], bf[4];
#pragma unroll
        for (int i = 0; i < 4; ++i) {
            const _Float16* p = A + (size_t)(mbase + i * 16 + lr) * DIM + k0 + half * 8;
            af[i] = load_frag_a(p);
            __builtin_prefetch(p + 32, 0, 0);
        }
#pragma unroll
        for (int j = 0; j < 4; ++j) {
            const _Float16* p = Bt + (size_t)(nbase + j * 16 + lr) * DIM + k0 + half * 8;
            bf[j] = load_frag_b(p);
            __builtin_prefetch(p + 32, 0, 0);
        }
#pragma unroll
        for (int i = 0; i < 4; ++i)
#pragma unroll
            for (int j = 0; j < 4; ++j) acc[i][j] = wmma16(af[i], bf[j], acc[i][j]);
    }

    // epilogue: bias + scatter to Q / K / V^T (f16)
#pragma unroll
    for (int i = 0; i < 4; ++i) {
#pragma unroll
        for (int j = 0; j < 4; ++j) {
            const int n  = nbase + j * 16 + lr;          // 0..3071
            const float bv = bias[n];
            const int which = n >> 10;                   // 0=Q 1=K 2=V
            const int nn = n & 1023;
            const int h  = nn >> 6;
            const int d  = nn & 63;
#pragma unroll
            for (int r = 0; r < 8; ++r) {
                const int t = mbase + i * 16 + r + 8 * half;   // token 0..4095
                const int b = t >> 11;
                const int s = t & 2047;
                const int bh = b * NHEAD + h;
                const _Float16 val = (_Float16)(acc[i][j][r] + bv);
                if (which == 0)
                    q16[((size_t)bh * SEQ + s) * HDIM + d] = val;
                else if (which == 1)
                    k16[((size_t)bh * SEQ + s) * HDIM + d] = val;
                else
                    vT[((size_t)bh * HDIM + d) * SEQ + s] = val;
            }
        }
    }
}

// ---------------- attention (flash, causal) ----------------
// One wave per (bh, 32-query tile). Output tile: 32 queries x 64 dims.
__global__ void __launch_bounds__(32)
k_attn(const _Float16* __restrict__ q16, const _Float16* __restrict__ k16,
       const _Float16* __restrict__ vT, _Float16* __restrict__ a16) {
    __shared__ __align__(16) _Float16 pbuf[2][16][32];

    const int lane = threadIdx.x & 31;
    const int half = lane >> 4;
    const int lr   = lane & 15;
    const int qbase = blockIdx.x * 32;     // 0..2016
    const int bh    = blockIdx.y;          // 0..31

    const _Float16* Q = q16 + (size_t)bh * SEQ * HDIM;
    const _Float16* K = k16 + (size_t)bh * SEQ * HDIM;
    const _Float16* V = vT  + (size_t)bh * HDIM * SEQ;

    // Q fragments (A operand): fixed for the whole key loop. qf[qtile][kchunk]
    Frag qf[2][2];
#pragma unroll
    for (int qt = 0; qt < 2; ++qt)
#pragma unroll
        for (int c = 0; c < 2; ++c)
            qf[qt][c] = load_frag_a(Q + (size_t)(qbase + qt * 16 + lr) * HDIM + c * 32 + half * 8);

    float m[2][8], l[2][8];
#pragma unroll
    for (int qt = 0; qt < 2; ++qt)
#pragma unroll
        for (int r = 0; r < 8; ++r) { m[qt][r] = -3.0e38f; l[qt][r] = 0.0f; }

    v8f o[2][4];
#pragma unroll
    for (int qt = 0; qt < 2; ++qt)
#pragma unroll
        for (int dt = 0; dt < 4; ++dt) o[qt][dt] = vzero();

    for (int kb = 0; kb <= qbase; kb += 32) {
        // K fragments (B operand: column = key): kf[key subtile][kchunk]
        Frag kf[2][2];
#pragma unroll
        for (int js = 0; js < 2; ++js)
#pragma unroll
            for (int c = 0; c < 2; ++c)
                kf[js][c] = load_frag_b(K + (size_t)(kb + js * 16 + lr) * HDIM + c * 32 + half * 8);

        // scores: s[qt][js] = Q_qt . K_js^T  (accumulate two hd-chunks)
        v8f sc[2][2];
#pragma unroll
        for (int qt = 0; qt < 2; ++qt)
#pragma unroll
            for (int js = 0; js < 2; ++js) {
                v8f t = wmma16(qf[qt][0], kf[js][0], vzero());
                sc[qt][js] = wmma16(qf[qt][1], kf[js][1], t);
            }

        // scale + causal mask
#pragma unroll
        for (int qt = 0; qt < 2; ++qt)
#pragma unroll
            for (int js = 0; js < 2; ++js) {
                const int key = kb + js * 16 + lr;
#pragma unroll
                for (int r = 0; r < 8; ++r) {
                    const int q = qbase + qt * 16 + r + 8 * half;
                    float v = sc[qt][js][r] * 0.125f;      // 1/sqrt(64)
                    sc[qt][js][r] = (key > q) ? -1.0e9f : v;
                }
            }

        // online softmax update + stage P into LDS (row-major 16x32 per qtile)
#pragma unroll
        for (int qt = 0; qt < 2; ++qt) {
            float corr[8];
#pragma unroll
            for (int r = 0; r < 8; ++r) {
                float tmax = rmax16(fmaxf(sc[qt][0][r], sc[qt][1][r]));
                float nm = fmaxf(m[qt][r], tmax);
                float p0 = __expf(sc[qt][0][r] - nm);
                float p1 = __expf(sc[qt][1][r] - nm);
                sc[qt][0][r] = p0;
                sc[qt][1][r] = p1;
                float rs = rsum16(p0 + p1);
                float cr = __expf(m[qt][r] - nm);
                l[qt][r] = l[qt][r] * cr + rs;
                m[qt][r] = nm;
                corr[r] = cr;
            }
#pragma unroll
            for (int dt = 0; dt < 4; ++dt)
#pragma unroll
                for (int r = 0; r < 8; ++r) o[qt][dt][r] *= corr[r];
#pragma unroll
            for (int js = 0; js < 2; ++js)
#pragma unroll
                for (int r = 0; r < 8; ++r)
                    pbuf[qt][r + 8 * half][js * 16 + lr] = (_Float16)sc[qt][js][r];
        }

        // wait for LDS stores, then read P back as the A operand of P.V
        asm volatile("s_wait_dscnt 0" ::: "memory");
        Frag pf[2];
#pragma unroll
        for (int qt = 0; qt < 2; ++qt)
            pf[qt] = load_frag_a(&pbuf[qt][lr][half * 8]);

        // V fragments (B operand from V^T: column d contiguous over keys)
        Frag vf[4];
#pragma unroll
        for (int dt = 0; dt < 4; ++dt)
            vf[dt] = load_frag_b(V + (size_t)(dt * 16 + lr) * SEQ + kb + half * 8);

#pragma unroll
        for (int qt = 0; qt < 2; ++qt)
#pragma unroll
            for (int dt = 0; dt < 4; ++dt)
                o[qt][dt] = wmma16(pf[qt], vf[dt], o[qt][dt]);
    }

    // normalize + write merged-head attn output [B*S][1024] (f16)
    const int b = bh >> 4;
    const int h = bh & 15;
#pragma unroll
    for (int qt = 0; qt < 2; ++qt)
#pragma unroll
        for (int dt = 0; dt < 4; ++dt) {
            const int d = dt * 16 + lr;
#pragma unroll
            for (int r = 0; r < 8; ++r) {
                const int s = qbase + qt * 16 + r + 8 * half;
                const float val = o[qt][dt][r] / l[qt][r];
                a16[((size_t)(b * SEQ + s)) * DIM + h * HDIM + d] = (_Float16)val;
            }
        }
}

// ---------------- output projection ----------------
// out[4096][1024] = a16 @ WprojT^T + b_proj, fp32 out. One wave per 64x64 tile.
__global__ void __launch_bounds__(32)
k_gemm_proj(const _Float16* __restrict__ A, const _Float16* __restrict__ Bt,
            const float* __restrict__ bias, float* __restrict__ out) {
    const int lane = threadIdx.x & 31;
    const int half = lane >> 4;
    const int lr   = lane & 15;
    const int nbase = blockIdx.x * 64;
    const int mbase = blockIdx.y * 64;

    v8f acc[4][4];
#pragma unroll
    for (int i = 0; i < 4; ++i)
#pragma unroll
        for (int j = 0; j < 4; ++j) acc[i][j] = vzero();

    for (int k0 = 0; k0 < DIM; k0 += 32) {
        Frag af[4], bf[4];
#pragma unroll
        for (int i = 0; i < 4; ++i) {
            const _Float16* p = A + (size_t)(mbase + i * 16 + lr) * DIM + k0 + half * 8;
            af[i] = load_frag_a(p);
            __builtin_prefetch(p + 32, 0, 0);
        }
#pragma unroll
        for (int j = 0; j < 4; ++j) {
            const _Float16* p = Bt + (size_t)(nbase + j * 16 + lr) * DIM + k0 + half * 8;
            bf[j] = load_frag_b(p);
        }
#pragma unroll
        for (int i = 0; i < 4; ++i)
#pragma unroll
            for (int j = 0; j < 4; ++j) acc[i][j] = wmma16(af[i], bf[j], acc[i][j]);
    }

#pragma unroll
    for (int i = 0; i < 4; ++i)
#pragma unroll
        for (int j = 0; j < 4; ++j) {
            const int n = nbase + j * 16 + lr;
            const float bv = bias[n];
#pragma unroll
            for (int r = 0; r < 8; ++r) {
                const int t = mbase + i * 16 + r + 8 * half;
                out[(size_t)t * DIM + n] = acc[i][j][r] + bv;
            }
        }
}

// ---------------- launch ----------------

extern "C" void kernel_launch(void* const* d_in, const int* in_sizes, int n_in,
                              void* d_out, int out_size, void* d_ws, size_t ws_size,
                              hipStream_t stream) {
    const float* x     = (const float*)d_in[0];   // [2,2048,1024]
    const float* Wqkv  = (const float*)d_in[1];   // [1024,3072]
    const float* bqkv  = (const float*)d_in[2];   // [3072]
    const float* Wproj = (const float*)d_in[3];   // [1024,1024]
    const float* bproj = (const float*)d_in[4];   // [1024]
    float* out = (float*)d_out;

    char* w = (char*)d_ws;                        // ~48 MB used
    _Float16* x16  = (_Float16*)w;  w += (size_t)TOKENS * DIM * 2;         // 8 MB
    _Float16* WqT  = (_Float16*)w;  w += (size_t)QKV_N * DIM * 2;          // 6 MB
    _Float16* WpT  = (_Float16*)w;  w += (size_t)DIM * DIM * 2;            // 2 MB
    _Float16* q16  = (_Float16*)w;  w += (size_t)BATCH * NHEAD * SEQ * HDIM * 2; // 8 MB
    _Float16* k16  = (_Float16*)w;  w += (size_t)BATCH * NHEAD * SEQ * HDIM * 2; // 8 MB
    _Float16* vT   = (_Float16*)w;  w += (size_t)BATCH * NHEAD * SEQ * HDIM * 2; // 8 MB
    _Float16* a16  = (_Float16*)w;  w += (size_t)TOKENS * DIM * 2;         // 8 MB

    {
        int n = TOKENS * DIM;
        k_cvt_f16<<<(n + 255) / 256, 256, 0, stream>>>(x, x16, n);
    }
    {
        int n = DIM * QKV_N;
        k_transpose_cvt<<<(n + 255) / 256, 256, 0, stream>>>(Wqkv, WqT, DIM, QKV_N);
    }
    {
        int n = DIM * DIM;
        k_transpose_cvt<<<(n + 255) / 256, 256, 0, stream>>>(Wproj, WpT, DIM, DIM);
    }

    k_gemm_qkv<<<dim3(QKV_N / 64, TOKENS / 64), 32, 0, stream>>>(x16, WqT, bqkv, q16, k16, vT);
    k_attn<<<dim3(SEQ / 32, BATCH * NHEAD), 32, 0, stream>>>(q16, k16, vT, a16);
    k_gemm_proj<<<dim3(DIM / 64, TOKENS / 64), 32, 0, stream>>>(a16, WpT, bproj, out);
}
